// AttentionHead_55353538511402
// MI455X (gfx1250) — matrix operand order repaired and checked by reading, not compiled, not run on hardware
//
#include <hip/hip_runtime.h>

// CDNA5 / gfx1250 wave32 WMMA fragment types
typedef __attribute__((ext_vector_type(16))) _Float16 v16h;
typedef __attribute__((ext_vector_type(8)))  float    v8f;
// b128 async payload type, matching the builtin's expected parameter type
typedef int v4i_ __attribute__((vector_size(16)));

#define WMMA_F16(A, B, C) \
  __builtin_amdgcn_wmma_f32_16x16x32_f16(false, (A), false, (B), (short)0, (C), false, false)

// ---------------------------------------------------------------------------
// Async global->LDS (gfx1250). Guarded: if the builtin is absent we fall back
// to a synchronous copy (still correct; waits become no-ops / asm).
// Probe-learned signature: (v4i AS1* src, v4i AS3* dst, imm offset, imm cpol).
// ---------------------------------------------------------------------------
#if defined(__has_builtin)
#if __has_builtin(__builtin_amdgcn_global_load_async_to_lds_b128)
#define HAVE_ASYNC_LDS 1
#endif
#endif
#ifndef HAVE_ASYNC_LDS
#define HAVE_ASYNC_LDS 0
#endif

__device__ __forceinline__ void async_cp16(_Float16* l, const _Float16* g) {
#if HAVE_ASYNC_LDS
  __builtin_amdgcn_global_load_async_to_lds_b128(
      (__attribute__((address_space(1))) v4i_*)g,
      (__attribute__((address_space(3))) v4i_*)l, 0, 0);
#else
  *(uint4*)l = *(const uint4*)g;
#endif
}

__device__ __forceinline__ void wait_async_le8() {
#if HAVE_ASYNC_LDS
#if __has_builtin(__builtin_amdgcn_s_wait_asynccnt)
  __builtin_amdgcn_s_wait_asynccnt(8);
#else
  asm volatile("s_wait_asynccnt 0x8" ::: "memory");
#endif
#endif
}
__device__ __forceinline__ void wait_async_0() {
#if HAVE_ASYNC_LDS
#if __has_builtin(__builtin_amdgcn_s_wait_asynccnt)
  __builtin_amdgcn_s_wait_asynccnt(0);
#else
  asm volatile("s_wait_asynccnt 0x0" ::: "memory");
#endif
#endif
}

// ---------------------------------------------------------------------------
// Fragment loaders, layouts used here (wave32), vectorized; A and B share one contiguous K order.
// A (16x32 f16): lane L -> row L&15; halves i = K base+i,
//                base = (L<16 ? 0 : 16).
// B (32x16 f16): lane L -> col L&15; halves i = K kofs+i, kofs=(L<16?0:16).
// C/D (16x16 f32): VGPR r -> row r + (L<16?0:8); col = L&15.
// ---------------------------------------------------------------------------
__device__ __forceinline__ v16h load_a_f16_vec(const _Float16* __restrict__ rows,
                                               int ld, int k0, int lane) {
  const _Float16* p = rows + (size_t)(lane & 15) * ld + k0 + ((lane < 16) ? 0 : 16);
  v16h a;
  *(uint4*)&a       = *(const uint4*)(p);       // K base+0..7
  *((uint4*)&a + 1) = *(const uint4*)(p + 8);   // K base+8..15
  return a;
}

__device__ __forceinline__ v16h load_a_f32_vec(const float* __restrict__ rows,
                                               int ld, int k0, int lane) {
  const float* p = rows + (size_t)(lane & 15) * ld + k0 + ((lane < 16) ? 0 : 16);
  float4 x0 = *(const float4*)(p);
  float4 x1 = *(const float4*)(p + 4);
  float4 x2 = *(const float4*)(p + 8);
  float4 x3 = *(const float4*)(p + 12);
  v16h a;
  a[0] = (_Float16)x0.x;  a[1] = (_Float16)x0.y;  a[2] = (_Float16)x0.z;  a[3] = (_Float16)x0.w;
  a[4] = (_Float16)x1.x;  a[5] = (_Float16)x1.y;  a[6] = (_Float16)x1.z;  a[7] = (_Float16)x1.w;
  a[8] = (_Float16)x2.x;  a[9] = (_Float16)x2.y;  a[10] = (_Float16)x2.z; a[11] = (_Float16)x2.w;
  a[12] = (_Float16)x3.x; a[13] = (_Float16)x3.y; a[14] = (_Float16)x3.z; a[15] = (_Float16)x3.w;
  return a;
}

// 16 contiguous halves (32 B) -> one B fragment's per-lane slice
__device__ __forceinline__ v16h load_contig16(const _Float16* p) {
  v16h b;
  *(uint4*)&b       = *(const uint4*)(p);
  *((uint4*)&b + 1) = *(const uint4*)(p + 8);
  return b;
}

// B from row-major [K x N] f32 source (used only by the weight pre-pack)
__device__ __forceinline__ v16h load_b_f32(const float* __restrict__ src,
                                           int ld, int k0, int n0, int lane) {
  const float* p = src + (size_t)(k0 + ((lane < 16) ? 0 : 16)) * ld + n0 + (lane & 15);
  v16h b;
#pragma unroll
  for (int v = 0; v < 8; ++v) {
    b[2 * v]     = (_Float16)p[(size_t)(2 * v) * ld];
    b[2 * v + 1] = (_Float16)p[(size_t)(2 * v + 1) * ld];
  }
  return b;
}

__device__ __forceinline__ v8f vzero() {
  return (v8f){0.f, 0.f, 0.f, 0.f, 0.f, 0.f, 0.f, 0.f};
}

// ---------------------------------------------------------------------------
// Kernel 0: pre-pack Wq/Wk/Wv into f16 B-fragment order.
// Fragment (mat, kc, f): 32 lanes x 16 halves, per-lane contiguous 32 B.
// 288 fragments total (3 mats x 24 K-steps x 4 N-tiles) = 288 KB, L2-resident.
// ---------------------------------------------------------------------------
__global__ __launch_bounds__(128) void weight_pack_kernel(
    const float* __restrict__ Wq, const float* __restrict__ Wk,
    const float* __restrict__ Wv, _Float16* __restrict__ wpack) {
  const int lane = threadIdx.x & 31;
  const int frag = blockIdx.x * 4 + (threadIdx.x >> 5);  // 0..287
  const int mat = frag / 96;
  const int rem = frag % 96;
  const int kc = rem >> 2;  // 0..23
  const int f  = rem & 3;   // 0..3
  const float* W = (mat == 0) ? Wq : (mat == 1) ? Wk : Wv;
  v16h b = load_b_f32(W, 64, kc * 32, f * 16, lane);
  *(v16h*)(wpack + ((size_t)frag * 32 + lane) * 16) = b;
}

// ---------------------------------------------------------------------------
// Kernel 1: fused QKV projection. One wave owns 16 rows and produces Q, K, V
// tiles from a single hidden read (12 accumulators, 12 WMMAs / K-step).
// Q,K stored row-major f16 (1/sqrt(D) folded into Q); V stored TRANSPOSED
// ([B][D][S] f16) so the attention kernel gets contiguous fragment loads.
// grid = 128, block = 256 (8 waves).
// ---------------------------------------------------------------------------
__global__ __launch_bounds__(256) void qkv_fused_kernel(
    const float* __restrict__ hidden, const _Float16* __restrict__ wpack,
    const float* __restrict__ bq, const float* __restrict__ bk,
    const float* __restrict__ bv, _Float16* __restrict__ qout,
    _Float16* __restrict__ kout, _Float16* __restrict__ vtout, int S) {
  const int E = 768, D = 64;
  const int lane = threadIdx.x & 31;
  const int wv   = threadIdx.x >> 5;
  const int row0 = (blockIdx.x * 8 + wv) * 16;
  const float* hrow = hidden + (size_t)row0 * E;

  v8f acc[3][4];
#pragma unroll
  for (int m = 0; m < 3; ++m)
#pragma unroll
    for (int f = 0; f < 4; ++f) acc[m][f] = vzero();

  for (int kc = 0; kc < E / 32; ++kc) {
    v16h a = load_a_f32_vec(hrow, E, kc * 32, lane);
#pragma unroll
    for (int m = 0; m < 3; ++m) {
#pragma unroll
      for (int f = 0; f < 4; ++f) {
        const _Float16* wp =
            wpack + ((size_t)((m * 24 + kc) * 4 + f) * 32 + lane) * 16;
        acc[m][f] = WMMA_F16(a, load_contig16(wp), acc[m][f]);
      }
    }
  }

  const int g = (lane < 16) ? 0 : 8;
  const int c = lane & 15;
  const int b  = row0 / S;
  const int sl = row0 - b * S + g;  // local seq position of first stored row
#pragma unroll
  for (int f = 0; f < 4; ++f) {
    const int d = f * 16 + c;
    const float biq = bq[d], bik = bk[d], biv = bv[d];
    union { uint4 u; _Float16 h[8]; } pk;
#pragma unroll
    for (int r = 0; r < 8; ++r) {
      qout[(size_t)(row0 + g + r) * D + d] = (_Float16)((acc[0][f][r] + biq) * 0.125f);
      kout[(size_t)(row0 + g + r) * D + d] = (_Float16)(acc[1][f][r] + bik);
      pk.h[r] = (_Float16)(acc[2][f][r] + biv);
    }
    // V^T: d-row is contiguous in s -> one 16 B store per fragment
    *(uint4*)(vtout + ((size_t)b * D + d) * S + sl) = pk.u;
  }
}

// ---------------------------------------------------------------------------
// Kernel 2: flash attention. 4 waves (4 query tiles, same batch) share
// double-buffered 64-key K and V^T tiles staged in LDS with async copies.
// Per iteration/wave: 8 WMMAs QK^T, online softmax, P relayout through a
// wave-private LDS tile, 8 WMMAs P@V. grid = 256, block = 128.
// ---------------------------------------------------------------------------
__global__ __launch_bounds__(128) void flash_attn_kernel(
    const _Float16* __restrict__ qbuf, const _Float16* __restrict__ kbuf,
    const _Float16* __restrict__ vtbuf, float* __restrict__ out, int S) {
  const int D = 64;
  __shared__ _Float16 kt[2][64 * 64];   // [key][dim]   8 KB x2
  __shared__ _Float16 vt[2][64 * 64];   // [dim][key]   8 KB x2
  __shared__ _Float16 pst[4][16 * 64];  // per-wave P staging, 8 KB

  const int t    = threadIdx.x;
  const int lane = t & 31;
  const int wv   = t >> 5;
  const int blk  = blockIdx.x;          // 32 blocks per batch
  const int b    = blk >> 5;
  const int s0   = ((blk & 31) * 4 + wv) * 16;

  const _Float16* qrow = qbuf + ((size_t)b * S + s0) * D;
  const _Float16* kgb  = kbuf + (size_t)b * S * D;      // row-major [S][D]
  const _Float16* vgb  = vtbuf + (size_t)b * D * S;     // transposed [D][S]

  const v16h qa0 = load_a_f16_vec(qrow, D, 0, lane);
  const v16h qa1 = load_a_f16_vec(qrow, D, 32, lane);

  float m[8], l[8];
  v8f acc[4];
#pragma unroll
  for (int r = 0; r < 8; ++r) { m[r] = -1e30f; l[r] = 0.f; }
#pragma unroll
  for (int f = 0; f < 4; ++f) acc[f] = vzero();

  const int g = (lane < 16) ? 0 : 8;
  const int c = lane & 15;
  const int NIT = S / 64;

  // --- tile fill: K tile is one contiguous 8 KB block; V^T tile is 64 rows
  // of 128 B. 128 threads x 16 B x 4 issues each => 8 async ops per wave.
  auto fill = [&](int buf, int key0) {
    _Float16* kl = kt[buf];
    _Float16* vl = vt[buf];
    const _Float16* kg = kgb + (size_t)key0 * D;
#pragma unroll
    for (int i = 0; i < 4; ++i) {
      int ch = t + i * 128;
      async_cp16(kl + ch * 8, kg + ch * 8);
    }
#pragma unroll
    for (int i = 0; i < 4; ++i) {
      int ch = t + i * 128;
      int row = ch >> 3, col = ch & 7;
      async_cp16(vl + ch * 8, vgb + (size_t)row * S + key0 + col * 8);
    }
  };

  fill(0, 0);
  for (int it = 0; it < NIT; ++it) {
    const int cur = it & 1;
    const bool more = (it + 1) < NIT;
    if (more) fill(1 - cur, (it + 1) * 64);
    if (more) wait_async_le8(); else wait_async_0();
    __syncthreads();  // all waves' tile data visible

    const _Float16* kl = kt[cur];
    const _Float16* vl = vt[cur];

    // S tile = Q (16x64) @ K_blk^T (64x64): 4 col-halves x 2 K-chunks.
    v8f sc[4];
#pragma unroll
    for (int h = 0; h < 4; ++h) {
      const _Float16* p = kl + (size_t)(h * 16 + c) * 64 + ((lane < 16) ? 0 : 16);
      sc[h] = WMMA_F16(qa0, load_contig16(p), vzero());
      sc[h] = WMMA_F16(qa1, load_contig16(p + 32), sc[h]);
    }

    // Online softmax (rows per-VGPR, cols across the 16-lane groups).
#pragma unroll
    for (int r = 0; r < 8; ++r) {
      float mx = fmaxf(fmaxf(sc[0][r], sc[1][r]), fmaxf(sc[2][r], sc[3][r]));
#pragma unroll
      for (int msk = 1; msk <= 8; msk <<= 1)
        mx = fmaxf(mx, __shfl_xor(mx, msk, 32));
      float mn  = fmaxf(m[r], mx);
      float rsc = __expf(m[r] - mn);
      m[r] = mn;
      float rs = 0.f;
#pragma unroll
      for (int h = 0; h < 4; ++h) {
        float p = __expf(sc[h][r] - mn);
        sc[h][r] = p;
        rs += p;
      }
#pragma unroll
      for (int msk = 1; msk <= 8; msk <<= 1)
        rs += __shfl_xor(rs, msk, 32);
      l[r] = l[r] * rsc + rs;
#pragma unroll
      for (int f = 0; f < 4; ++f) acc[f][r] *= rsc;
    }

    // Relayout P (C layout) -> A layout via wave-private LDS tile.
#pragma unroll
    for (int h = 0; h < 4; ++h)
#pragma unroll
      for (int r = 0; r < 8; ++r)
        pst[wv][(g + r) * 64 + h * 16 + c] = (_Float16)sc[h][r];
    asm volatile("" ::: "memory");  // keep IR ordering; DS is in-order per wave
    v16h pa0 = load_a_f16_vec(pst[wv], 64, 0, lane);
    v16h pa1 = load_a_f16_vec(pst[wv], 64, 32, lane);

    // O += P (16x64) @ V_blk (64x64), V^T rows give contiguous loads.
#pragma unroll
    for (int f = 0; f < 4; ++f) {
      const _Float16* p = vl + (size_t)(f * 16 + c) * 64 + ((lane < 16) ? 0 : 16);
      acc[f] = WMMA_F16(pa0, load_contig16(p), acc[f]);
      acc[f] = WMMA_F16(pa1, load_contig16(p + 32), acc[f]);
    }
    __syncthreads();  // done reading before next overwrite of this buffer
  }

  // Epilogue: normalize, store f32 [B, S, D].
#pragma unroll
  for (int r = 0; r < 8; ++r) {
    float inv = 1.0f / l[r];
#pragma unroll
    for (int f = 0; f < 4; ++f)
      out[((size_t)b * S + s0 + g + r) * D + f * 16 + c] = acc[f][r] * inv;
  }
}

// ---------------------------------------------------------------------------
// Host-side launch
// ---------------------------------------------------------------------------
extern "C" void kernel_launch(void* const* d_in, const int* in_sizes, int n_in,
                              void* d_out, int out_size, void* d_ws, size_t ws_size,
                              hipStream_t stream) {
  const int B = 8, S = 2048, E = 768, D = 64;
  const int M = B * S;

  const float* hidden = (const float*)d_in[0];
  const float* Wq = (const float*)d_in[1];
  const float* bq = (const float*)d_in[2];
  const float* Wk = (const float*)d_in[3];
  const float* bk = (const float*)d_in[4];
  const float* Wv = (const float*)d_in[5];
  const float* bv = (const float*)d_in[6];

  _Float16* qbuf  = (_Float16*)d_ws;
  _Float16* kbuf  = qbuf + (size_t)M * D;
  _Float16* vtbuf = kbuf + (size_t)M * D;            // [B][D][S]
  _Float16* wpack = vtbuf + (size_t)M * D;           // 288 KB fragment pack

  weight_pack_kernel<<<dim3(288 / 4), 128, 0, stream>>>(Wq, Wk, Wv, wpack);

  qkv_fused_kernel<<<dim3(M / (16 * 8)), 256, 0, stream>>>(
      hidden, wpack, bq, bk, bv, qbuf, kbuf, vtbuf, S);

  flash_attn_kernel<<<dim3((B * (S / 16)) / 4), 128, 0, stream>>>(
      qbuf, kbuf, vtbuf, (float*)d_out, S);
}
